// PerceiverAttention_31636729103168
// MI455X (gfx1250) — matrix operand order repaired and checked by reading, not compiled, not run on hardware
//
#include <hip/hip_runtime.h>

typedef _Float16 half_t;
typedef __attribute__((ext_vector_type(2)))  _Float16 v2h;
typedef __attribute__((ext_vector_type(4)))  _Float16 v4h;
typedef __attribute__((ext_vector_type(8)))  _Float16 v8h;
typedef __attribute__((ext_vector_type(16))) _Float16 v16h;
typedef __attribute__((ext_vector_type(8)))  float    v8f;
typedef __attribute__((vector_size(16)))     int      v4i_vs;   // pointee type of async-LDS builtins

#define SHUF16(lo, hi) __builtin_shufflevector((lo), (hi), 0,1,2,3,4,5,6,7,8,9,10,11,12,13,14,15)
#define WMMA_F16(a, b, c) __builtin_amdgcn_wmma_f32_16x16x32_f16(false, (a), false, (b), (short)0, (c), false, false)

#define B_SZ 4
#define M_SZ 8192
#define L_SZ 64
#define D_MODEL 1024
#define D_HEAD 64
#define NHEADS 8
#define IN_DIM 512
#define SROWS (M_SZ + L_SZ)      // 8256 rows per batch in kv-input / kv
#define NCHUNK (SROWS / 64)      // 129 key chunks of 64
#define NSPLIT 8                 // flash split-j factor

// ---- CDNA5 async global->LDS staging (guarded; falls back to sync copies) ----
#define AS1 __attribute__((address_space(1)))
#define AS3 __attribute__((address_space(3)))
#if defined(__gfx1250__) && __has_builtin(__builtin_amdgcn_global_load_async_to_lds_b128) && \
    __has_builtin(__builtin_amdgcn_s_wait_asynccnt)
#define HAVE_ASYNC_LDS 1
#else
#define HAVE_ASYNC_LDS 0
#endif

static __device__ __forceinline__ void lds_copy16(const half_t* g, half_t* l) {
#if HAVE_ASYNC_LDS
  __builtin_amdgcn_global_load_async_to_lds_b128((AS1 v4i_vs*)g, (AS3 v4i_vs*)l, 0, 0);
#else
  *(v8h*)l = *(const v8h*)g;
#endif
}
static __device__ __forceinline__ void lds_copy_fence() {
#if HAVE_ASYNC_LDS
  __builtin_amdgcn_s_wait_asynccnt(0);
#endif
}

static __device__ __forceinline__ v8f vzero8f() {
  v8f z = {0.f, 0.f, 0.f, 0.f, 0.f, 0.f, 0.f, 0.f};
  return z;
}

// ---------------------------------------------------------------- f32 -> f16
__global__ void f32_to_f16_kernel(const float* __restrict__ s, half_t* __restrict__ d, int n) {
  int i = blockIdx.x * 256 + threadIdx.x;
  if (i < n) d[i] = (half_t)s[i];
}

// ------------------------------------------------- fused LayerNorm -> f16 row
__global__ __launch_bounds__(256) void ln_to_f16_kernel(
    const float* __restrict__ in, const float* __restrict__ g, const float* __restrict__ be,
    half_t* __restrict__ out, int rows_per_batch, int out_batch_stride, int out_row_offset) {
  const int row = blockIdx.x;
  const int b = row / rows_per_batch;
  const int i = row % rows_per_batch;
  const float4* src = (const float4*)(in + (size_t)row * D_MODEL);
  float4 v = src[threadIdx.x];
  float vv[4] = {v.x, v.y, v.z, v.w};
  float s = vv[0] + vv[1] + vv[2] + vv[3];
  float q = vv[0]*vv[0] + vv[1]*vv[1] + vv[2]*vv[2] + vv[3]*vv[3];
  #pragma unroll
  for (int off = 16; off >= 1; off >>= 1) {
    s += __shfl_xor(s, off);
    q += __shfl_xor(q, off);
  }
  __shared__ float ss[8], sq[8];
  const int w = threadIdx.x >> 5;
  if ((threadIdx.x & 31) == 0) { ss[w] = s; sq[w] = q; }
  __syncthreads();
  s = 0.f; q = 0.f;
  #pragma unroll
  for (int k = 0; k < 8; ++k) { s += ss[k]; q += sq[k]; }
  const float mu = s * (1.0f / D_MODEL);
  const float var = q * (1.0f / D_MODEL) - mu * mu;
  const float rstd = rsqrtf(var + 1e-5f);
  half_t* dst = out + ((size_t)b * out_batch_stride + out_row_offset + i) * D_MODEL;
  const int c = threadIdx.x * 4;
  v4h o;
  #pragma unroll
  for (int k = 0; k < 4; ++k)
    o[k] = (half_t)((vv[k] - mu) * rstd * g[c + k] + be[c + k]);
  *(v4h*)(dst + c) = o;
}

// ------------------------------------------------------- tiled WMMA f16 GEMM
// C[M,N] = A[M,K] @ B[K,N]. Block tile 128x128, 8 waves, wave = 32x64.
// Double-buffered LDS; A staged with async-to-LDS, B staged with a paired
// (k,k+1) transpose so stores are ds_store_b32.
template <bool OUT32>
__global__ __launch_bounds__(256) void wmma_gemm_kernel(
    const half_t* __restrict__ A, const half_t* __restrict__ B,
    half_t* __restrict__ C16, float* __restrict__ C32,
    int Mrows, int N, int K, int rpg, int gstride_rows, int rowoff) {
  __shared__ __align__(16) half_t As[2][128 * 32];   // [row][k] row-major
  __shared__ __align__(16) half_t Bt[2][128 * 32];   // [n][k]  (B transposed)
  const int tid = threadIdx.x, lane = tid & 31, wave = tid >> 5;
  const int wm = wave & 3, wn = wave >> 2;
  const int l15 = lane & 15;
  const int lhi = (lane >= 16);
  const int rowbase = blockIdx.x * 128;
  const int colbase = blockIdx.y * 128;

  v8f acc[2][4];
  #pragma unroll
  for (int mt = 0; mt < 2; ++mt)
    #pragma unroll
    for (int nt = 0; nt < 4; ++nt) acc[mt][nt] = vzero8f();

  // A staging: thread t -> row (t>>1), half-range ((t&1)*16)
  const int arow_l = tid >> 1;
  const int ac0 = (tid & 1) * 16;
  const int r = rowbase + arow_l;
  const size_t gr = (size_t)(r / rpg) * (size_t)gstride_rows + (size_t)rowoff + (size_t)(r % rpg);
  const half_t* aptr = A + gr * (size_t)K + ac0;
  // B staging: thread t -> k-pair (t>>4)*2, col-range ((t&15)*8)
  const int bn0 = (tid & 15) * 8;
  const int bk = (tid >> 4) * 2;
  const half_t* bptr = B + (size_t)bk * N + colbase + bn0;

  const int nk = K / 32;
  // prologue: stage tile 0 into buffer 0
  {
    lds_copy16(aptr, &As[0][arow_l * 32 + ac0]);
    lds_copy16(aptr + 8, &As[0][arow_l * 32 + ac0 + 8]);
    v8h b0 = *(const v8h*)(bptr);
    v8h b1 = *(const v8h*)(bptr + N);
    #pragma unroll
    for (int i = 0; i < 8; ++i) {
      v2h p = {b0[i], b1[i]};
      *(v2h*)(&Bt[0][(bn0 + i) * 32 + bk]) = p;
    }
  }
  lds_copy_fence();
  __syncthreads();

  for (int t = 0; t < nk; ++t) {
    const int buf = t & 1;
    // stage next tile into the other buffer while computing this one
    if (t + 1 < nk) {
      const int kt = (t + 1) * 32;
      lds_copy16(aptr + kt, &As[buf ^ 1][arow_l * 32 + ac0]);
      lds_copy16(aptr + kt + 8, &As[buf ^ 1][arow_l * 32 + ac0 + 8]);
      v8h b0 = *(const v8h*)(bptr + (size_t)kt * N);
      v8h b1 = *(const v8h*)(bptr + (size_t)(kt + 1) * N);
      #pragma unroll
      for (int i = 0; i < 8; ++i) {
        v2h p = {b0[i], b1[i]};
        *(v2h*)(&Bt[buf ^ 1][(bn0 + i) * 32 + bk]) = p;
      }
    }
    // compute on current buffer
    v16h af[2];
    #pragma unroll
    for (int mt = 0; mt < 2; ++mt) {
      const half_t* ap = &As[buf][(wm * 32 + mt * 16 + l15) * 32 + (lhi ? 8 : 0)];
      v8h lo = *(const v8h*)ap;
      v8h hi = *(const v8h*)(ap + 16);
      af[mt] = SHUF16(lo, hi);
    }
    #pragma unroll
    for (int nt = 0; nt < 4; ++nt) {
      const half_t* bp = &Bt[buf][(wn * 64 + nt * 16 + l15) * 32 + (lhi ? 8 : 0)];
      v8h lo = *(const v8h*)bp;
      v8h hi = *(const v8h*)(bp + 16);
      v16h bf = SHUF16(lo, hi);
      acc[0][nt] = WMMA_F16(af[0], bf, acc[0][nt]);
      acc[1][nt] = WMMA_F16(af[1], bf, acc[1][nt]);
    }
    lds_copy_fence();   // my async stores into buf^1 are done
    __syncthreads();    // everyone done reading buf / writing buf^1
  }

  #pragma unroll
  for (int mt = 0; mt < 2; ++mt)
    #pragma unroll
    for (int nt = 0; nt < 4; ++nt)
      #pragma unroll
      for (int rr = 0; rr < 8; ++rr) {
        const int row = rowbase + wm * 32 + mt * 16 + rr + (lhi ? 8 : 0);
        const int col = colbase + wn * 64 + nt * 16 + l15;
        if constexpr (OUT32) C32[(size_t)row * N + col] = acc[mt][nt][rr];
        else                 C16[(size_t)row * N + col] = (half_t)acc[mt][nt][rr];
      }
  (void)Mrows;
}

// ---------------------------------------------------- flash attention, split-j
// grid (NSPLIT, H, B); 128 threads = 4 waves; wave w owns q-rows [16w,16w+16).
// 64-column key chunks: one softmax/lane-reduction round per 64 columns.
__global__ __launch_bounds__(128) void flash_attn_kernel(
    const half_t* __restrict__ q16, const half_t* __restrict__ kv16,
    float* __restrict__ Opart, float* __restrict__ Mpart, float* __restrict__ Lpart) {
  const int sp = blockIdx.x, h = blockIdx.y, b = blockIdx.z;
  const int tid = threadIdx.x, lane = tid & 31, wave = tid >> 5;
  const int l15 = lane & 15;
  const int lhi = (lane >= 16);
  const int kb0 = lhi ? 8 : 0;
  const int rh = lhi ? 8 : 0;

  __shared__ __align__(16) half_t Klds[64 * 64];    // [j][dh]  == Bt layout for Q@K^T
  __shared__ __align__(16) half_t Vt[64 * 64];      // [dh][j]  == Bt layout for P@V
  __shared__ __align__(16) half_t Plds[4 * 16 * 64];

  // preload Q A-fragments (dh split in two K=32 steps)
  const half_t* qrow = q16 + (size_t)(b * 64 + wave * 16 + l15) * IN_DIM + h * D_HEAD;
  v16h aQ[2];
  #pragma unroll
  for (int s = 0; s < 2; ++s) {
    v8h lo = *(const v8h*)(qrow + 32 * s + kb0);
    v8h hi = *(const v8h*)(qrow + 32 * s + kb0 + 16);
    aQ[s] = SHUF16(lo, hi);
  }

  float mrow[8], lrow[8];
  v8f accO[4];
  #pragma unroll
  for (int r = 0; r < 8; ++r) { mrow[r] = -1e30f; lrow[r] = 0.f; }
  #pragma unroll
  for (int nt = 0; nt < 4; ++nt) accO[nt] = vzero8f();

  // K staging: thread t -> key row (t>>1), 32 halves at ((t&1)*32)
  const int kjr = tid >> 1;
  const int kc0 = (tid & 1) * 32;
  // V staging: thread t -> key-row pair 2*(t>>2), 16 dh values at ((t&3)*16)
  const int vjp = (tid >> 2) * 2;
  const int vd0 = (tid & 3) * 16;

  for (int c = sp; c < NCHUNK; c += NSPLIT) {
    const int jb = c * 64;
    const half_t* krow = kv16 + (size_t)(b * SROWS + jb + kjr) * D_MODEL + h * D_HEAD + kc0;
    lds_copy16(krow,      Klds + kjr * 64 + kc0);
    lds_copy16(krow + 8,  Klds + kjr * 64 + kc0 + 8);
    lds_copy16(krow + 16, Klds + kjr * 64 + kc0 + 16);
    lds_copy16(krow + 24, Klds + kjr * 64 + kc0 + 24);
    const half_t* vrow0 = kv16 + (size_t)(b * SROWS + jb + vjp) * D_MODEL + IN_DIM + h * D_HEAD + vd0;
    v8h v0lo = *(const v8h*)vrow0;
    v8h v0hi = *(const v8h*)(vrow0 + 8);
    v8h v1lo = *(const v8h*)(vrow0 + D_MODEL);
    v8h v1hi = *(const v8h*)(vrow0 + D_MODEL + 8);
    #pragma unroll
    for (int i = 0; i < 8; ++i) {
      v2h plo = {v0lo[i], v1lo[i]};
      v2h phi = {v0hi[i], v1hi[i]};
      *(v2h*)(Vt + (vd0 + i) * 64 + vjp)     = plo;
      *(v2h*)(Vt + (vd0 + 8 + i) * 64 + vjp) = phi;
    }
    lds_copy_fence();
    __syncthreads();

    // scores S = (q*8) @ K^T  -> four 16x16 fragments (cols 16f..16f+15)
    v8f sfrag[4];
    #pragma unroll
    for (int f = 0; f < 4; ++f) {
      v8f a = vzero8f();
      #pragma unroll
      for (int s = 0; s < 2; ++s) {
        const half_t* bp = Klds + (f * 16 + l15) * 64 + s * 32 + (lhi ? 8 : 0);
        v8h lo = *(const v8h*)bp;
        v8h hi = *(const v8h*)(bp + 16);
        v16h bk = SHUF16(lo, hi);
        a = WMMA_F16(aQ[s], bk, a);
      }
      sfrag[f] = a * 8.0f;   // reference: q / dh^-0.5 == q * sqrt(64)
    }

    // one online-softmax round per 64 columns
    #pragma unroll
    for (int r = 0; r < 8; ++r) {
      float mx = fmaxf(fmaxf(sfrag[0][r], sfrag[1][r]), fmaxf(sfrag[2][r], sfrag[3][r]));
      mx = fmaxf(mx, __shfl_xor(mx, 1, 16));
      mx = fmaxf(mx, __shfl_xor(mx, 2, 16));
      mx = fmaxf(mx, __shfl_xor(mx, 4, 16));
      mx = fmaxf(mx, __shfl_xor(mx, 8, 16));
      const float mnew  = fmaxf(mrow[r], mx);
      const float alpha = __expf(mrow[r] - mnew);
      mrow[r] = mnew;
      float rs = 0.f;
      #pragma unroll
      for (int f = 0; f < 4; ++f) {
        const float p = __expf(sfrag[f][r] - mnew);
        sfrag[f][r] = p;
        rs += p;
      }
      rs += __shfl_xor(rs, 1, 16);
      rs += __shfl_xor(rs, 2, 16);
      rs += __shfl_xor(rs, 4, 16);
      rs += __shfl_xor(rs, 8, 16);
      lrow[r] = lrow[r] * alpha + rs;
      #pragma unroll
      for (int nt = 0; nt < 4; ++nt) accO[nt][r] *= alpha;
    }

    // C-layout -> A-layout for P via this wave's private LDS tile
    half_t* pb = Plds + wave * (16 * 64);
    #pragma unroll
    for (int f = 0; f < 4; ++f)
      #pragma unroll
      for (int r = 0; r < 8; ++r)
        pb[(r + rh) * 64 + f * 16 + l15] = (half_t)sfrag[f][r];
    v16h aP[2];
    #pragma unroll
    for (int s = 0; s < 2; ++s) {
      v8h plo = *(const v8h*)(pb + l15 * 64 + 32 * s + kb0);
      v8h phi = *(const v8h*)(pb + l15 * 64 + 32 * s + kb0 + 16);
      aP[s] = SHUF16(plo, phi);
    }

    // O += P @ V  (K = 64 in two steps)
    #pragma unroll
    for (int nt = 0; nt < 4; ++nt)
      #pragma unroll
      for (int s = 0; s < 2; ++s) {
        const half_t* vp = Vt + (nt * 16 + l15) * 64 + 32 * s + (lhi ? 8 : 0);
        v8h lo = *(const v8h*)vp;
        v8h hi = *(const v8h*)(vp + 16);
        v16h bv = SHUF16(lo, hi);
        accO[nt] = WMMA_F16(aP[s], bv, accO[nt]);
      }
    __syncthreads();
  }

  // emit partials
  const int P = (b * NHEADS + h) * NSPLIT + sp;
  float* Op = Opart + (size_t)P * (64 * 64);
  #pragma unroll
  for (int nt = 0; nt < 4; ++nt)
    #pragma unroll
    for (int r = 0; r < 8; ++r) {
      const int qi = wave * 16 + r + rh;
      const int d  = nt * 16 + l15;
      Op[qi * 64 + d] = accO[nt][r];
    }
  if (l15 == 0) {
    #pragma unroll
    for (int r = 0; r < 8; ++r) {
      const int qi = wave * 16 + r + rh;
      Mpart[P * 64 + qi] = mrow[r];
      Lpart[P * 64 + qi] = lrow[r];
    }
  }
}

// --------------------------------------------- combine split-j flash partials
__global__ __launch_bounds__(256) void attn_combine_kernel(
    const float* __restrict__ Opart, const float* __restrict__ Mp, const float* __restrict__ Lp,
    half_t* __restrict__ attn16) {
  const int h = blockIdx.x, b = blockIdx.y;
  const int bh = b * NHEADS + h;
  __shared__ float sL[64];
  __shared__ float sW[NSPLIT][64];
  if (threadIdx.x < 64) {
    const int row = threadIdx.x;
    float M = -1e30f;
    #pragma unroll
    for (int s = 0; s < NSPLIT; ++s) M = fmaxf(M, Mp[(bh * NSPLIT + s) * 64 + row]);
    float L = 0.f;
    #pragma unroll
    for (int s = 0; s < NSPLIT; ++s) {
      const float w = __expf(Mp[(bh * NSPLIT + s) * 64 + row] - M);
      sW[s][row] = w;
      L += Lp[(bh * NSPLIT + s) * 64 + row] * w;
    }
    sL[row] = L;
  }
  __syncthreads();
  const int row = threadIdx.x >> 2;
  const int d0 = (threadIdx.x & 3) * 16;
  const float invL = 1.f / sL[row];
  for (int d = d0; d < d0 + 16; ++d) {
    float a = 0.f;
    #pragma unroll
    for (int s = 0; s < NSPLIT; ++s)
      a += sW[s][row] * Opart[((size_t)(bh * NSPLIT + s)) * (64 * 64) + row * 64 + d];
    attn16[(size_t)(b * 64 + row) * IN_DIM + h * D_HEAD + d] = (half_t)(a * invL);
  }
}

// ---------------------------------------------------------------------- host
extern "C" void kernel_launch(void* const* d_in, const int* in_sizes, int n_in,
                              void* d_out, int out_size, void* d_ws, size_t ws_size,
                              hipStream_t stream) {
  (void)in_sizes; (void)n_in; (void)out_size; (void)ws_size;
  const float* x       = (const float*)d_in[0];
  const float* latents = (const float*)d_in[1];
  const float* Wq      = (const float*)d_in[2];
  const float* Wkv     = (const float*)d_in[3];
  const float* Wout    = (const float*)d_in[4];
  const float* gm      = (const float*)d_in[5];
  const float* bm      = (const float*)d_in[6];
  const float* gl      = (const float*)d_in[7];
  const float* bl      = (const float*)d_in[8];
  float* out = (float*)d_out;

  char* ws = (char*)d_ws;
  size_t off = 0;
  auto carve = [&](size_t bytes) -> char* {
    off = (off + 255) & ~(size_t)255;
    char* p = ws + off;
    off += bytes;
    return p;
  };
  const size_t kvin_elems = (size_t)B_SZ * SROWS * D_MODEL;
  half_t* kvin16 = (half_t*)carve(kvin_elems * 2);
  half_t* kv16   = (half_t*)carve(kvin_elems * 2);
  half_t* q16    = (half_t*)carve((size_t)B_SZ * L_SZ * IN_DIM * 2);
  half_t* attn16 = (half_t*)carve((size_t)B_SZ * L_SZ * IN_DIM * 2);
  half_t* Wq16   = (half_t*)carve((size_t)D_MODEL * IN_DIM * 2);
  half_t* Wkv16  = (half_t*)carve((size_t)D_MODEL * D_MODEL * 2);
  half_t* Wout16 = (half_t*)carve((size_t)IN_DIM * D_MODEL * 2);
  float*  Opart  = (float*)carve((size_t)B_SZ * NHEADS * NSPLIT * 64 * 64 * 4);
  float*  Mpart  = (float*)carve((size_t)B_SZ * NHEADS * NSPLIT * 64 * 4);
  float*  Lpart  = (float*)carve((size_t)B_SZ * NHEADS * NSPLIT * 64 * 4);

  f32_to_f16_kernel<<<(D_MODEL * IN_DIM + 255) / 256, 256, 0, stream>>>(Wq, Wq16, D_MODEL * IN_DIM);
  f32_to_f16_kernel<<<(D_MODEL * D_MODEL + 255) / 256, 256, 0, stream>>>(Wkv, Wkv16, D_MODEL * D_MODEL);
  f32_to_f16_kernel<<<(IN_DIM * D_MODEL + 255) / 256, 256, 0, stream>>>(Wout, Wout16, IN_DIM * D_MODEL);

  ln_to_f16_kernel<<<B_SZ * M_SZ, 256, 0, stream>>>(x, gm, bm, kvin16, M_SZ, SROWS, 0);
  ln_to_f16_kernel<<<B_SZ * L_SZ, 256, 0, stream>>>(latents, gl, bl, kvin16, L_SZ, SROWS, M_SZ);

  wmma_gemm_kernel<false><<<dim3(B_SZ * SROWS / 128, D_MODEL / 128), 256, 0, stream>>>(
      kvin16, Wkv16, kv16, nullptr, B_SZ * SROWS, D_MODEL, D_MODEL, 1 << 30, 0, 0);

  wmma_gemm_kernel<false><<<dim3((B_SZ * L_SZ) / 128, IN_DIM / 128), 256, 0, stream>>>(
      kvin16, Wq16, q16, nullptr, B_SZ * L_SZ, IN_DIM, D_MODEL, L_SZ, SROWS, M_SZ);

  flash_attn_kernel<<<dim3(NSPLIT, NHEADS, B_SZ), 128, 0, stream>>>(q16, kv16, Opart, Mpart, Lpart);
  attn_combine_kernel<<<dim3(NHEADS, B_SZ), 256, 0, stream>>>(Opart, Mpart, Lpart, attn16);

  wmma_gemm_kernel<true><<<dim3((B_SZ * L_SZ) / 128, D_MODEL / 128), 256, 0, stream>>>(
      attn16, Wout16, nullptr, out, B_SZ * L_SZ, D_MODEL, IN_DIM, 1 << 30, 0, 0);
}
